// DotProductAttention_89404039234059
// MI455X (gfx1250) — matrix operand order repaired and checked by reading, not compiled, not run on hardware
//
#include <hip/hip_runtime.h>

// DotProductAttention, flash-attention style, bf16 WMMA on gfx1250 (wave32).
// B=2, H=8, S=4096, D=64. Pass 1 converts K/V to bf16 WMMA-fragment layout in
// d_ws; pass 2 is the flash loop fetching fragments as b128 loads.

#define SEQ  4096
#define HD   64
#define NWAVES 8
#define QROWS (NWAVES * 16)
#define NBH   16                      // B*H

typedef __attribute__((ext_vector_type(8)))  float  v8f;
typedef __attribute__((ext_vector_type(16))) __bf16 v16bf;

union Frag {
    unsigned int u[8];
    uint4 q4[2];
    v16bf v;
};

// pack two floats to bf16 (round-half-up) into one dword: a->[15:0], b->[31:16]
__device__ __forceinline__ unsigned int pack2bf(float a, float b) {
    const unsigned int ua = __float_as_uint(a) + 0x8000u;
    const unsigned int ub = __float_as_uint(b) + 0x8000u;
    // v_perm_b32: concat = {S0,S1}, sel byte picks concat byte
    return __builtin_amdgcn_perm(ua, ub, 0x03020706u);
}

// ws layout:
//   wsK: [bh][kg=key/16][ch=dchunk][lane][2 x uint4]   (A-operand frags of K)
//   wsV: [bh][kbi=key/32][n=dtile ][lane][2 x uint4]   (B-operand frags of V)
#define WSK_U4_PER_BH ((size_t)256 * 2 * 32 * 2)
#define WSV_U4_PER_BH ((size_t)128 * 4 * 32 * 2)

__global__ __launch_bounds__(NWAVES * 32)
void kv_convert_kernel(const float* __restrict__ K, const float* __restrict__ V,
                       uint4* __restrict__ wsK, uint4* __restrict__ wsV) {
    const int lane  = threadIdx.x & 31;
    const int wave  = threadIdx.x >> 5;
    const int l16   = lane & 15;
    const int lhalf = lane >> 4;

    const int w   = blockIdx.x * NWAVES + wave;   // 0 .. NBH*128-1
    const int bh  = w >> 7;
    const int kbi = w & 127;
    const int kb  = kbi * 32;

    const float* Kb = K + (size_t)bh * SEQ * HD;
    const float* Vb = V + (size_t)bh * SEQ * HD;

    // K as A-operand frags: M = key, K-dim = d.
#pragma unroll
    for (int t = 0; t < 2; ++t) {
        const int kg = kbi * 2 + t;
        const float* krow = Kb + (size_t)(kb + t * 16 + l16) * HD;
#pragma unroll
        for (int ch = 0; ch < 2; ++ch) {
            Frag ka;
#pragma unroll
            for (int v = 0; v < 8; ++v) {
                // A layout (16-bit 16x32): K = 16*(v>=4) + 8*(lane/16) + 2*(v&3), +1
                const int d = ch * 32 + ((v >= 4) ? 16 : 0) + lhalf * 8 + (v & 3) * 2;
                const float2 f = *(const float2*)(krow + d);
                ka.u[v] = pack2bf(f.x, f.y);
            }
            uint4* dst = wsK + ((size_t)((bh * 256 + kg) * 2 + ch) * 32 + lane) * 2;
            dst[0] = ka.q4[0];
            dst[1] = ka.q4[1];
        }
    }

    // V as B-operand frags: K-dim = key, N = d.
#pragma unroll
    for (int n = 0; n < 4; ++n) {
        Frag vb;
        const int d = n * 16 + l16;
#pragma unroll
        for (int v = 0; v < 8; ++v) {
            // B layout (16-bit 32x16): K = 16*(v>=4) + 8*(lane/16) + 2*(v&3), +1
            const int key = kb + ((v >= 4) ? 16 : 0) + lhalf * 8 + (v & 3) * 2;
            const float f0 = Vb[(size_t)key * HD + d];
            const float f1 = Vb[(size_t)(key + 1) * HD + d];
            vb.u[v] = pack2bf(f0, f1);
        }
        uint4* dst = wsV + ((size_t)((bh * 128 + kbi) * 4 + n) * 32 + lane) * 2;
        dst[0] = vb.q4[0];
        dst[1] = vb.q4[1];
    }
}

template <bool USE_WS>
__global__ __launch_bounds__(NWAVES * 32)
void DotProductAttention_89404039234059_kernel(const float* __restrict__ Q,
                                               const float* __restrict__ K,
                                               const float* __restrict__ V,
                                               const unsigned char* __restrict__ Msk,
                                               float* __restrict__ Out,
                                               const uint4* __restrict__ wsK,
                                               const uint4* __restrict__ wsV) {
    const int lane  = threadIdx.x & 31;
    const int wave  = threadIdx.x >> 5;
    const int l16   = lane & 15;
    const int lhalf = lane >> 4;

    const int qtiles = SEQ / QROWS;           // 32
    const int qtile  = blockIdx.x % qtiles;
    const int bh     = blockIdx.x / qtiles;
    const int qbase  = qtile * QROWS + wave * 16;

    // 1/sqrt(64) * log2(e), folded into the Q bf16 conversion
    const float SCL  = 0.125f * 1.44269504088896340f;
    const float NEGI = -1.0e30f;              // masked sentinel (exp2 underflows to 0)

    const float* Qb = Q + (size_t)bh * SEQ * HD;
    const float* Kb = K + (size_t)bh * SEQ * HD;
    const float* Vb = V + (size_t)bh * SEQ * HD;
    float*       Ob = Out + (size_t)bh * SEQ * HD;
    const uint4* kf = wsK + (size_t)bh * WSK_U4_PER_BH;
    const uint4* vf = wsV + (size_t)bh * WSV_U4_PER_BH;

    // ---- Q (pre-scaled) as B-operand of S^T = K_tile x Q^T ----
    Frag qf[2];
    {
        const float* qrow = Qb + (size_t)(qbase + l16) * HD;
#pragma unroll
        for (int c = 0; c < 2; ++c) {
#pragma unroll
            for (int v = 0; v < 8; ++v) {
                const int d = c * 32 + ((v >= 4) ? 16 : 0) + lhalf * 8 + (v & 3) * 2;
                const float2 f = *(const float2*)(qrow + d);
                qf[c].u[v] = pack2bf(f.x * SCL, f.y * SCL);
            }
        }
    }

    v8f o[4];
#pragma unroll
    for (int n = 0; n < 4; ++n)
#pragma unroll
        for (int i = 0; i < 8; ++i) o[n][i] = 0.0f;

    float m = NEGI;      // running max for q = qbase + lane%16 (replicated halves)
    float l = 0.0f;      // running sum

    const unsigned char* mrow = Msk + (size_t)(qbase + l16) * SEQ;

#pragma unroll 1
    for (int kb = 0; kb < SEQ; kb += 32) {
        // ---- S^T tiles: st[t] covers keys kb+16t .. kb+16t+15 ----
        v8f st[2];
#pragma unroll
        for (int t = 0; t < 2; ++t) {
            v8f c;
#pragma unroll
            for (int i = 0; i < 8; ++i) c[i] = 0.0f;
#pragma unroll
            for (int ch = 0; ch < 2; ++ch) {
                Frag ka;
                if constexpr (USE_WS) {
                    const int kg = (kb >> 4) + t;
                    const uint4* p = kf + ((size_t)((kg * 2 + ch) * 32 + lane)) * 2;
                    ka.q4[0] = p[0];
                    ka.q4[1] = p[1];
                } else {
                    const float* krow = Kb + (size_t)(kb + t * 16 + l16) * HD;
#pragma unroll
                    for (int v = 0; v < 8; ++v) {
                        const int d = ch * 32 + ((v >= 4) ? 16 : 0) + lhalf * 8 + (v & 3) * 2;
                        const float2 f = *(const float2*)(krow + d);
                        ka.u[v] = pack2bf(f.x, f.y);
                    }
                }
                c = __builtin_amdgcn_wmma_f32_16x16x32_bf16(
                        false, ka.v, false, qf[ch].v, (short)0, c, false, false);
            }
            st[t] = c;
        }

        // ---- mask: 8 consecutive key bytes per tile for this lane ----
        uint2 mw[2];
#pragma unroll
        for (int t = 0; t < 2; ++t)
            mw[t] = *(const uint2*)(mrow + kb + t * 16 + lhalf * 8);

        // ---- online softmax (state indexed by q = lane%16) ----
        // se = s + (maskbyte - 1) * 1e30 : byte==1 -> s exactly; byte==0 -> -1e30
        // (byte extract + uitofp combines to v_cvt_f32_ubyteJ; no cmp/cndmask)
        float se[2][8];
        float tmax = NEGI;
#pragma unroll
        for (int t = 0; t < 2; ++t)
#pragma unroll
            for (int v = 0; v < 8; ++v) {
                const unsigned int w32 = (v < 4) ? mw[t].x : mw[t].y;
                const float fb = (float)((w32 >> (8 * (v & 3))) & 0xffu);
                se[t][v] = fmaf(fb - 1.0f, 1.0e30f, st[t][v]);
                tmax = fmaxf(tmax, se[t][v]);
            }
        tmax = fmaxf(tmax, __shfl_xor(tmax, 16, 32));
        const float mnew = fmaxf(m, tmax);

        float p[2][8];
        float rs = 0.0f;
#pragma unroll
        for (int t = 0; t < 2; ++t)
#pragma unroll
            for (int v = 0; v < 8; ++v) {
                const float pe = exp2f(se[t][v] - mnew);   // masked -> exact 0
                p[t][v] = pe;
                rs += pe;
            }
        rs += __shfl_xor(rs, 16, 32);

        // Rescale O only when some row's max actually moved (wave-uniform branch;
        // expected ~H(128) ~= 5 of 128 iterations). EXEC stays all-ones.
        if (__any(mnew > m)) {
            const float alpha = exp2f(m - mnew);   // m==NEGI -> 0, resets any prefix
            float areg[8];
#pragma unroll
            for (int i = 0; i < 8; ++i) areg[i] = __shfl(alpha, i + 8 * lhalf, 32);
#pragma unroll
            for (int n = 0; n < 4; ++n)
#pragma unroll
                for (int i = 0; i < 8; ++i) o[n][i] *= areg[i];
            l = l * alpha + rs;
            m = mnew;
        } else {
            l += rs;
        }

        // ---- P^T (C layout) -> P as A-operand: register repack ----
        Frag pa;
#pragma unroll
        for (int vv = 0; vv < 4; ++vv) {
            pa.u[vv]     = pack2bf(p[0][2 * vv], p[0][2 * vv + 1]);
            pa.u[4 + vv] = pack2bf(p[1][2 * vv], p[1][2 * vv + 1]);
        }

        // ---- O += P(16x32) x V(32x16) per d-tile ----
#pragma unroll
        for (int n = 0; n < 4; ++n) {
            Frag vb;
            if constexpr (USE_WS) {
                const uint4* p2 = vf + ((size_t)(((kb >> 5) * 4 + n) * 32 + lane)) * 2;
                vb.q4[0] = p2[0];
                vb.q4[1] = p2[1];
            } else {
                const int d = n * 16 + l16;
#pragma unroll
                for (int v = 0; v < 8; ++v) {
                    const int key = kb + ((v >= 4) ? 16 : 0) + lhalf * 8 + (v & 3) * 2;
                    const float f0 = Vb[(size_t)key * HD + d];
                    const float f1 = Vb[(size_t)(key + 1) * HD + d];
                    vb.u[v] = pack2bf(f0, f1);
                }
            }
            o[n] = __builtin_amdgcn_wmma_f32_16x16x32_bf16(
                       false, pa.v, false, vb.v, (short)0, o[n], false, false);
        }

        // prefetch next block's fragments (global_prefetch_b8)
        if (kb + 32 < SEQ) {
            if constexpr (USE_WS) {
                __builtin_prefetch(kf + ((size_t)((((kb >> 4) + 2) * 2) * 32 + lane)) * 2, 0, 3);
                __builtin_prefetch(vf + ((size_t)((((kb >> 5) + 1) * 4) * 32 + lane)) * 2, 0, 3);
            } else {
                __builtin_prefetch(Kb + (size_t)(kb + 32 + lane) * HD, 0, 3);
                __builtin_prefetch(Vb + (size_t)(kb + 32 + lane) * HD, 0, 3);
            }
        }
    }

    // ---- epilogue ----
    const float linv = (l > 0.0f) ? (1.0f / l) : 0.0f;
    float lreg[8];
#pragma unroll
    for (int i = 0; i < 8; ++i) lreg[i] = __shfl(linv, i + 8 * lhalf, 32);
#pragma unroll
    for (int n = 0; n < 4; ++n) {
        const int d = n * 16 + l16;
#pragma unroll
        for (int i = 0; i < 8; ++i) {
            const int qq = qbase + i + 8 * lhalf;
            Ob[(size_t)qq * HD + d] = o[n][i] * lreg[i];
        }
    }
}

extern "C" void kernel_launch(void* const* d_in, const int* in_sizes, int n_in,
                              void* d_out, int out_size, void* d_ws, size_t ws_size,
                              hipStream_t stream) {
    // setup_inputs order: queries, keys, values, d_k(scalar, unused), mask(bool bytes)
    const float* q = (const float*)d_in[0];
    const float* k = (const float*)d_in[1];
    const float* v = (const float*)d_in[2];
    const unsigned char* msk = (const unsigned char*)d_in[4];
    float* out = (float*)d_out;

    const int grid = NBH * (SEQ / QROWS);     // 16 * 32 = 512 workgroups

    const size_t needK = WSK_U4_PER_BH * NBH * sizeof(uint4);  // 8 MiB
    const size_t needV = WSV_U4_PER_BH * NBH * sizeof(uint4);  // 8 MiB
    if (ws_size >= needK + needV) {
        uint4* wsK = (uint4*)d_ws;
        uint4* wsV = (uint4*)((char*)d_ws + needK);
        kv_convert_kernel<<<NBH * 128 / NWAVES, NWAVES * 32, 0, stream>>>(k, v, wsK, wsV);
        DotProductAttention_89404039234059_kernel<true>
            <<<grid, NWAVES * 32, 0, stream>>>(q, k, v, msk, out, wsK, wsV);
    } else {
        DotProductAttention_89404039234059_kernel<false>
            <<<grid, NWAVES * 32, 0, stream>>>(q, k, v, msk, out, nullptr, nullptr);
    }
    (void)in_sizes; (void)n_in; (void)out_size;
}
